// CausalAnalysisPredictor_39986145525795
// MI455X (gfx1250) — matrix operand order repaired and checked by reading, not compiled, not run on hardware
//
#include <hip/hip_runtime.h>
#include <hip/hip_bf16.h>

// ---------------------------------------------------------------------------
// CausalAnalysisPredictor on MI455X (gfx1250, wave32, WMMA).
// ~231 GFLOP vs ~330 MB HBM -> compute bound on the 16384x{1024,512}x4096
// GEMMs; path: v_wmma_f32_16x16x32_bf16 (f32 accumulate).
// Round 3: (1) 32-row M-blocking per wave in the two heavy kernels so each
// B-fragment (2x global_load_b128) feeds two WMMAs -> loads/WMMA ~1.5 and
// half the L2 weight traffic. (2) pairwise f32->bf16 packing into u32
// (v_perm_b32) instead of per-half inserts. Weights remain pre-transposed
// bf16 (N-major, zero-padded); post_ctx (268 MB) is never materialized.
// ---------------------------------------------------------------------------

typedef __bf16 bf16_t;
typedef __attribute__((ext_vector_type(16))) __bf16 v16bf;
typedef __attribute__((ext_vector_type(8)))  __bf16 v8bf;
typedef __attribute__((ext_vector_type(8)))  float  v8f;
typedef __attribute__((ext_vector_type(4)))  float  v4f;
typedef __attribute__((ext_vector_type(8)))  unsigned vu8;
typedef __attribute__((ext_vector_type(4)))  unsigned vu4;

#define N_OBJ 4096
#define N_REL 16384
#define HH    512
#define PP    4096
#define NOC   151
#define NRC   51
#define NRC_PAD 64

// float -> bf16 (RNE), scalar form
__device__ __forceinline__ bf16_t f2bf(float x) {
  unsigned u = __builtin_bit_cast(unsigned, x);
  u += 0x7FFFu + ((u >> 16) & 1u);
  unsigned short s = (unsigned short)(u >> 16);
  return __builtin_bit_cast(bf16_t, s);
}

// two floats -> packed bf16x2 in a u32 (RNE); tail is a v_perm_b32 pattern
__device__ __forceinline__ unsigned bfpack2(float lo, float hi) {
  unsigned a = __builtin_bit_cast(unsigned, lo);
  unsigned b = __builtin_bit_cast(unsigned, hi);
  a += 0x7FFFu + ((a >> 16) & 1u);
  b += 0x7FFFu + ((b >> 16) & 1u);
  return (a >> 16) | (b & 0xFFFF0000u);
}

__device__ __forceinline__ v16bf cat16(v8bf lo, v8bf hi) {
  return __builtin_shufflevector(lo, hi, 0, 1, 2, 3, 4, 5, 6, 7,
                                 8, 9, 10, 11, 12, 13, 14, 15);
}

__device__ __forceinline__ v8f wmma_bf16(v16bf a, v16bf b, v8f c) {
  return __builtin_amdgcn_wmma_f32_16x16x32_bf16(
      false, a, false, b, (short)0, c, false, false);
}

// A-fragment (16x32 bf16): lane = row M, halves 0..7 = K off..off+7,
// halves 8..15 = K off+16..off+23, off = kb + (lane>=16 ? 8 : 0).
__device__ __forceinline__ v16bf a_frag_bf16(const bf16_t* __restrict__ rowbase,
                                             int kb, int lane) {
  const int off = kb + ((lane & 16) ? 8 : 0);
  v8bf r0 = *(const v8bf*)(rowbase + off);
  v8bf r1 = *(const v8bf*)(rowbase + off + 16);
  return cat16(r0, r1);
}

// Same fragment from an fp32 row: 4x float4 loads, pairwise pack (8 perms).
__device__ __forceinline__ v16bf a_frag_f32(const float* __restrict__ rowbase,
                                            int kb, int lane) {
  const int off = kb + ((lane & 16) ? 8 : 0);
  v4f f0 = *(const v4f*)(rowbase + off);
  v4f f1 = *(const v4f*)(rowbase + off + 4);
  v4f f2 = *(const v4f*)(rowbase + off + 16);
  v4f f3 = *(const v4f*)(rowbase + off + 20);
  vu8 u;
  u[0] = bfpack2(f0[0], f0[1]); u[1] = bfpack2(f0[2], f0[3]);
  u[2] = bfpack2(f1[0], f1[1]); u[3] = bfpack2(f1[2], f1[3]);
  u[4] = bfpack2(f2[0], f2[1]); u[5] = bfpack2(f2[2], f2[3]);
  u[6] = bfpack2(f3[0], f3[1]); u[7] = bfpack2(f3[2], f3[3]);
  return __builtin_bit_cast(v16bf, u);
}

// B-fragment (32x16 bf16) from pre-transposed bf16 weight Wt[N][K] (ldk):
// lane = column N, halves 0..7 = K off..off+7, 8..15 = K off+16..off+23,
// off = kb + (lane>=16 ? 8 : 0). Two b128 loads, no conversion.
__device__ __forceinline__ v16bf b_frag_t(const bf16_t* __restrict__ Wt, int ldk,
                                          int kb, int n0, int lane) {
  const bf16_t* p = Wt + (size_t)(n0 + (lane & 15)) * ldk + kb +
                    ((lane & 16) ? 8 : 0);
  v8bf lo = *(const v8bf*)(p);
  v8bf hi = *(const v8bf*)(p + 16);
  return cat16(lo, hi);
}

// --------------------------------------------------------------------------
// Prepass: Wt[n][k] (bf16, rows n >= N zero-filled up to Npad) = W[k][n] (f32)
// --------------------------------------------------------------------------
__global__ void __launch_bounds__(256)
k_w_transpose(const float* __restrict__ W, bf16_t* __restrict__ Wt,
              int K, int N, int Npad) {
  __shared__ float tile[32][33];
  const int k0 = blockIdx.x * 32;
  const int n0 = blockIdx.y * 32;
#pragma unroll
  for (int yy = threadIdx.y; yy < 32; yy += 8) {
    const int k = k0 + yy, n = n0 + threadIdx.x;
    tile[yy][threadIdx.x] = (n < N && k < K) ? W[(size_t)k * N + n] : 0.f;
  }
  __syncthreads();
#pragma unroll
  for (int yy = threadIdx.y; yy < 32; yy += 8) {
    const int n = n0 + yy, k = k0 + threadIdx.x;
    if (n < Npad) Wt[(size_t)n * K + k] = f2bf(tile[threadIdx.x][yy]);
  }
}

// --------------------------------------------------------------------------
// Kernel A: edge_rep(4096x1024, bf16) = edge_ctx @ W_post_emb + b
// grid (64, 16), block 128 (wave = 16 rows x 64 cols)
// --------------------------------------------------------------------------
__global__ void __launch_bounds__(128)
k_edge_emb(const float* __restrict__ X, const bf16_t* __restrict__ Wt,
           const float* __restrict__ bias, bf16_t* __restrict__ Y) {
  const int lane  = threadIdx.x & 31;
  const int wave  = threadIdx.x >> 5;
  const int mBase = blockIdx.x * 64 + wave * 16;
  const int n0    = blockIdx.y * 64;
  const float* rowbase = X + (size_t)(mBase + (lane & 15)) * HH;
  v8f acc[4] = {};
  for (int kb = 0; kb < HH; kb += 32) {
    v16bf a = a_frag_f32(rowbase, kb, lane);
#pragma unroll
    for (int j = 0; j < 4; ++j) {
      v16bf b = b_frag_t(Wt, HH, kb, n0 + 16 * j, lane);
      acc[j] = wmma_bf16(a, b, acc[j]);
    }
  }
  const int mhi = (lane >> 4) & 1;
#pragma unroll
  for (int j = 0; j < 4; ++j) {
    const int n = n0 + 16 * j + (lane & 15);
    const float bv = bias[n];
#pragma unroll
    for (int i = 0; i < 8; ++i) {
      const int m = mBase + 8 * mhi + i;
      Y[(size_t)m * (2 * HH) + n] = f2bf(acc[j][i] + bv);
    }
  }
}

// --------------------------------------------------------------------------
// Kernel B: spt_h(16384x512, bf16) = relu(pair_bbox @ W_spt1 + b); K=32.
// grid (256, 8), block 128
// --------------------------------------------------------------------------
__global__ void __launch_bounds__(128)
k_spt1(const float* __restrict__ BB, const bf16_t* __restrict__ Wt,
       const float* __restrict__ bias, bf16_t* __restrict__ Y) {
  const int lane  = threadIdx.x & 31;
  const int wave  = threadIdx.x >> 5;
  const int mBase = blockIdx.x * 64 + wave * 16;
  const int n0    = blockIdx.y * 64;
  const float* rowbase = BB + (size_t)(mBase + (lane & 15)) * 32;
  v8f acc[4] = {};
  v16bf a = a_frag_f32(rowbase, 0, lane);
#pragma unroll
  for (int j = 0; j < 4; ++j) {
    v16bf b = b_frag_t(Wt, 32, 0, n0 + 16 * j, lane);
    acc[j] = wmma_bf16(a, b, acc[j]);
  }
  const int mhi = (lane >> 4) & 1;
#pragma unroll
  for (int j = 0; j < 4; ++j) {
    const int n = n0 + 16 * j + (lane & 15);
    const float bv = bias[n];
#pragma unroll
    for (int i = 0; i < 8; ++i) {
      const int m = mBase + 8 * mhi + i;
      float v = acc[j][i] + bv;
      Y[(size_t)m * HH + n] = f2bf(v > 0.f ? v : 0.f);
    }
  }
}

// --------------------------------------------------------------------------
// Kernel C (fused core): wave owns 32 rows; loop over P in 64-col tiles;
// pc = relu(gather(edge_rep) @ W_post_cat + b), sp = relu(spt_h @ W_spt2 + b);
// stage pc*sp (bf16) through LDS; contract with W_ctx / W_gate into
// persistent 51-wide f32 accumulators. grid 128, block 128.
// --------------------------------------------------------------------------
__global__ void __launch_bounds__(128)
k_main(const bf16_t* __restrict__ edge, const int* __restrict__ pair_idx,
       const bf16_t* __restrict__ spt,
       const bf16_t* __restrict__ WtPc, const float* __restrict__ bpc,
       const bf16_t* __restrict__ WtS2, const float* __restrict__ bs2,
       const bf16_t* __restrict__ WtC,  const float* __restrict__ bc,
       const bf16_t* __restrict__ WtG,  const float* __restrict__ bg,
       float* __restrict__ outCtx, float* __restrict__ outGate) {
  __shared__ __align__(16) bf16_t lds[4][64 * 32];  // per-wave 32x64, [n][m]
  const int lane  = threadIdx.x & 31;
  const int wave  = threadIdx.x >> 5;
  const int rBase = blockIdx.x * 128 + wave * 32;
  const int r0    = rBase + (lane & 15);
  const int r1    = r0 + 16;

  // ctx_rep[r,k] = edge_rep[k<512 ? head : tail, k]  (same column offset)
  const int h0 = pair_idx[2 * r0 + 0], t0 = pair_idx[2 * r0 + 1];
  const int h1 = pair_idx[2 * r1 + 0], t1 = pair_idx[2 * r1 + 1];
  const bf16_t* ctxLo0 = edge + (size_t)h0 * (2 * HH);
  const bf16_t* ctxHi0 = edge + (size_t)t0 * (2 * HH);
  const bf16_t* ctxLo1 = edge + (size_t)h1 * (2 * HH);
  const bf16_t* ctxHi1 = edge + (size_t)t1 * (2 * HH);
  const bf16_t* sptRow0 = spt + (size_t)r0 * HH;
  const bf16_t* sptRow1 = spt + (size_t)r1 * HH;
  bf16_t* myLds = lds[wave];
  const int mhi = (lane >> 4) & 1;
  const int mm  = lane & 15;

  v8f cCtx[2][4] = {}, cGate[2][4] = {};

  for (int c0 = 0; c0 < PP; c0 += 64) {
    // ---- pc = ctx_rep @ W_post_cat (K = 1024); B shared by 2 M-subtiles ----
    v8f pcA[2][4] = {};
    for (int kb = 0; kb < 2 * HH; kb += 32) {
      const bf16_t* rb0 = (kb < HH) ? ctxLo0 : ctxHi0;
      const bf16_t* rb1 = (kb < HH) ? ctxLo1 : ctxHi1;
      v16bf a0 = a_frag_bf16(rb0, kb, lane);
      v16bf a1 = a_frag_bf16(rb1, kb, lane);
#pragma unroll
      for (int j = 0; j < 4; ++j) {
        v16bf b = b_frag_t(WtPc, 2 * HH, kb, c0 + 16 * j, lane);
        pcA[0][j] = wmma_bf16(a0, b, pcA[0][j]);
        pcA[1][j] = wmma_bf16(a1, b, pcA[1][j]);
      }
    }
    // ---- sp = spt_h @ W_spt2 (K = 512) ----
    v8f spA[2][4] = {};
    for (int kb = 0; kb < HH; kb += 32) {
      v16bf a0 = a_frag_bf16(sptRow0, kb, lane);
      v16bf a1 = a_frag_bf16(sptRow1, kb, lane);
#pragma unroll
      for (int j = 0; j < 4; ++j) {
        v16bf b = b_frag_t(WtS2, HH, kb, c0 + 16 * j, lane);
        spA[0][j] = wmma_bf16(a0, b, spA[0][j]);
        spA[1][j] = wmma_bf16(a1, b, spA[1][j]);
      }
    }
    // ---- combine: relu(pc+b1)*relu(sp+b2); packed 16B LDS stores ----
#pragma unroll
    for (int s = 0; s < 2; ++s) {
#pragma unroll
      for (int j = 0; j < 4; ++j) {
        const int nloc = 16 * j + mm;
        const int n = c0 + nloc;
        const float b1 = bpc[n], b2 = bs2[n];
        vu4 pk;
#pragma unroll
        for (int ii = 0; ii < 4; ++ii) {
          float p0 = pcA[s][j][2 * ii]     + b1; p0 = p0 > 0.f ? p0 : 0.f;
          float p1 = pcA[s][j][2 * ii + 1] + b1; p1 = p1 > 0.f ? p1 : 0.f;
          float q0 = spA[s][j][2 * ii]     + b2; q0 = q0 > 0.f ? q0 : 0.f;
          float q1 = spA[s][j][2 * ii + 1] + b2; q1 = q1 > 0.f ? q1 : 0.f;
          pk[ii] = bfpack2(p0 * q0, p1 * q1);
        }
        *(vu4*)(myLds + nloc * 32 + 16 * s + 8 * mhi) = pk;
      }
    }
    // ---- second-level GEMM: tile(32x64) @ W_ctx / W_gate (K = 64) ----
#pragma unroll
    for (int kk = 0; kk < 64; kk += 32) {
      const int koff = kk + ((lane & 16) ? 8 : 0);
      v16bf a2[2];
#pragma unroll
      for (int s = 0; s < 2; ++s) {
        const int mcol = 16 * s + mm;
#pragma unroll
        for (int i = 0; i < 8; ++i) a2[s][i]     = myLds[(koff + i) * 32 + mcol];
#pragma unroll
        for (int i = 0; i < 8; ++i) a2[s][8 + i] = myLds[(koff + 16 + i) * 32 + mcol];
      }
#pragma unroll
      for (int j = 0; j < 4; ++j) {
        v16bf bC = b_frag_t(WtC, PP, c0 + kk, 16 * j, lane);
        cCtx[0][j] = wmma_bf16(a2[0], bC, cCtx[0][j]);
        cCtx[1][j] = wmma_bf16(a2[1], bC, cCtx[1][j]);
        v16bf bG = b_frag_t(WtG, PP, c0 + kk, 16 * j, lane);
        cGate[0][j] = wmma_bf16(a2[0], bG, cGate[0][j]);
        cGate[1][j] = wmma_bf16(a2[1], bG, cGate[1][j]);
      }
    }
  }
  // ---- write ctx/gate dists (with bias), padded ld = 64 ----
#pragma unroll
  for (int s = 0; s < 2; ++s) {
#pragma unroll
    for (int j = 0; j < 4; ++j) {
      const int n = 16 * j + mm;
      const float bcv = (n < NRC) ? bc[n] : 0.f;
      const float bgv = (n < NRC) ? bg[n] : 0.f;
#pragma unroll
      for (int i = 0; i < 8; ++i) {
        const int m = rBase + 16 * s + 8 * mhi + i;
        outCtx [(size_t)m * NRC_PAD + n] = cCtx[s][j][i]  + bcv;
        outGate[(size_t)m * NRC_PAD + n] = cGate[s][j][i] + bgv;
      }
    }
  }
}

// --------------------------------------------------------------------------
// Kernel D: vis = union_features @ W_vis (streams 268 MB once), fused with
// freq gather + sigmoid gating + final output. grid 128, block 128;
// wave owns 32 rows so each B-fragment feeds two WMMAs.
// --------------------------------------------------------------------------
__global__ void __launch_bounds__(128)
k_vis_final(const float* __restrict__ U, const bf16_t* __restrict__ WtV,
            const float* __restrict__ bv, const int* __restrict__ pair_pred,
            const float* __restrict__ freq, const float* __restrict__ ctxD,
            const float* __restrict__ gateD, float* __restrict__ out) {
  const int lane  = threadIdx.x & 31;
  const int wave  = threadIdx.x >> 5;
  const int rBase = blockIdx.x * 128 + wave * 32;
  const float* row0 = U + (size_t)(rBase + (lane & 15)) * PP;
  const float* row1 = row0 + (size_t)16 * PP;
  v8f acc[2][4] = {};
  for (int kb = 0; kb < PP; kb += 32) {
    v16bf a0 = a_frag_f32(row0, kb, lane);
    v16bf a1 = a_frag_f32(row1, kb, lane);
#pragma unroll
    for (int j = 0; j < 4; ++j) {
      v16bf b = b_frag_t(WtV, PP, kb, 16 * j, lane);
      acc[0][j] = wmma_bf16(a0, b, acc[0][j]);
      acc[1][j] = wmma_bf16(a1, b, acc[1][j]);
    }
  }
  const int mhi = (lane >> 4) & 1;
#pragma unroll
  for (int s = 0; s < 2; ++s) {
#pragma unroll
    for (int j = 0; j < 4; ++j) {
      const int n = 16 * j + (lane & 15);
      if (n < NRC) {
        const float bvv = bv[n];
#pragma unroll
        for (int i = 0; i < 8; ++i) {
          const int m = rBase + 16 * s + 8 * mhi + i;
          const int p0 = pair_pred[2 * m], p1 = pair_pred[2 * m + 1];
          const float frq = freq[(size_t)(p0 * NOC + p1) * NRC + n];
          const float vis = acc[s][j][i] + bvv;
          const float sg  = vis + frq + gateD[(size_t)m * NRC_PAD + n];
          const float sig = 1.0f / (1.0f + __expf(-sg));
          out[(size_t)m * NRC + n] = ctxD[(size_t)m * NRC_PAD + n] * sig;
        }
      }
    }
  }
}

extern "C" void kernel_launch(void* const* d_in, const int* in_sizes, int n_in,
                              void* d_out, int out_size, void* d_ws, size_t ws_size,
                              hipStream_t stream) {
  const float* edge_ctx   = (const float*)d_in[0];
  const int*   pair_idx   = (const int*)  d_in[1];
  const int*   pair_pred  = (const int*)  d_in[2];
  const float* pair_bbox  = (const float*)d_in[3];
  const float* unionf     = (const float*)d_in[4];
  const float* W_post_emb = (const float*)d_in[5];
  const float* b_post_emb = (const float*)d_in[6];
  const float* W_post_cat = (const float*)d_in[7];
  const float* b_post_cat = (const float*)d_in[8];
  const float* W_spt1     = (const float*)d_in[9];
  const float* b_spt1     = (const float*)d_in[10];
  const float* W_spt2     = (const float*)d_in[11];
  const float* b_spt2     = (const float*)d_in[12];
  const float* W_ctx      = (const float*)d_in[13];
  const float* b_ctx      = (const float*)d_in[14];
  const float* W_vis      = (const float*)d_in[15];
  const float* b_vis      = (const float*)d_in[16];
  const float* W_gate     = (const float*)d_in[17];
  const float* b_gate     = (const float*)d_in[18];
  const float* freq_table = (const float*)d_in[19];

  char* ws = (char*)d_ws;
  const size_t MB = 1u << 20;
  bf16_t* ws_edge = (bf16_t*)(ws);            //  8 MB: 4096x1024 bf16
  bf16_t* ws_spt  = (bf16_t*)(ws +  8 * MB);  // 16 MB: 16384x512 bf16
  float*  ws_ctx  = (float*) (ws + 24 * MB);  //  4 MB: 16384x64 f32
  float*  ws_gate = (float*) (ws + 28 * MB);  //  4 MB: 16384x64 f32
  bf16_t* wt_pc   = (bf16_t*)(ws + 32 * MB);  //  8 MB: [4096][1024]
  bf16_t* wt_s2   = (bf16_t*)(ws + 40 * MB);  //  4 MB: [4096][512]
  bf16_t* wt_emb  = (bf16_t*)(ws + 44 * MB);  //  1 MB: [1024][512]
  bf16_t* wt_s1   = (bf16_t*)(ws + 45 * MB);  // 32 KB: [512][32]
  bf16_t* wt_c    = (bf16_t*)(ws + 46 * MB);  // 512 KB: [64][4096]
  bf16_t* wt_g    = (bf16_t*)(ws + 47 * MB);  // 512 KB: [64][4096]
  bf16_t* wt_v    = (bf16_t*)(ws + 48 * MB);  // 512 KB: [64][4096]
  float*  out     = (float*)d_out;

  // ---- one-shot weight convert+transpose (bf16, N-major, zero-padded) ----
  k_w_transpose<<<dim3(HH / 32, (2 * HH) / 32), dim3(32, 8), 0, stream>>>(
      W_post_emb, wt_emb, HH, 2 * HH, 2 * HH);
  k_w_transpose<<<dim3(1, HH / 32), dim3(32, 8), 0, stream>>>(
      W_spt1, wt_s1, 32, HH, HH);
  k_w_transpose<<<dim3((2 * HH) / 32, PP / 32), dim3(32, 8), 0, stream>>>(
      W_post_cat, wt_pc, 2 * HH, PP, PP);
  k_w_transpose<<<dim3(HH / 32, PP / 32), dim3(32, 8), 0, stream>>>(
      W_spt2, wt_s2, HH, PP, PP);
  k_w_transpose<<<dim3(PP / 32, NRC_PAD / 32), dim3(32, 8), 0, stream>>>(
      W_ctx, wt_c, PP, NRC, NRC_PAD);
  k_w_transpose<<<dim3(PP / 32, NRC_PAD / 32), dim3(32, 8), 0, stream>>>(
      W_gate, wt_g, PP, NRC, NRC_PAD);
  k_w_transpose<<<dim3(PP / 32, NRC_PAD / 32), dim3(32, 8), 0, stream>>>(
      W_vis, wt_v, PP, NRC, NRC_PAD);

  // ---- pipeline ----
  k_edge_emb<<<dim3(N_OBJ / 64, (2 * HH) / 64), 128, 0, stream>>>(
      edge_ctx, wt_emb, b_post_emb, ws_edge);
  k_spt1<<<dim3(N_REL / 64, HH / 64), 128, 0, stream>>>(
      pair_bbox, wt_s1, b_spt1, ws_spt);
  k_main<<<dim3(N_REL / 128), 128, 0, stream>>>(
      ws_edge, pair_idx, ws_spt, wt_pc, b_post_cat, wt_s2, b_spt2,
      wt_c, b_ctx, wt_g, b_gate, ws_ctx, ws_gate);
  k_vis_final<<<dim3(N_REL / 128), 128, 0, stream>>>(
      unionf, wt_v, b_vis, pair_pred, freq_table, ws_ctx, ws_gate, out);
}
